// Block_7584912244938
// MI455X (gfx1250) — matrix operand order repaired and checked by reading, not compiled, not run on hardware
//
#include <hip/hip_runtime.h>
#include <cstdint>
#include <cstddef>

typedef __attribute__((ext_vector_type(16))) _Float16 v16h;
typedef __attribute__((ext_vector_type(8)))  _Float16 h8;
typedef __attribute__((ext_vector_type(8)))  float    v8f;

#define HEADS 16
#define DIMH  64
#define SPAD_MAX 224

// ---------------------------------------------------------------- helpers

__device__ inline v8f vzero8() {
  v8f z;
#pragma unroll
  for (int i = 0; i < 8; ++i) z[i] = 0.0f;
  return z;
}

__device__ inline v8f wmma_f16(v16h a, v16h b, v8f c) {
  return __builtin_amdgcn_wmma_f32_16x16x32_f16(false, a, false, b, (short)0, c,
                                                false, false);
}

__device__ inline float gelu_f(float x) {
  const float c = 0.7978845608028654f;  // sqrt(2/pi)
  float t = tanhf(c * (x + 0.044715f * x * x * x));
  return 0.5f * x * (1.0f + t);
}

// A fragment (16x32 f16, MxK): lane l: m = l&15; lanes 0-15 hold K 0-7 & 16-23,
// lanes 16-31 hold K 8-15 & 24-31 (2 packed f16 per VGPR).
__device__ inline v16h load_a_frag(const _Float16* A, int lda, int row0, int k0,
                                   int lane) {
  int m  = lane & 15;
  int hi = lane >> 4;
  const _Float16* p = A + (size_t)(row0 + m) * lda + k0 + hi * 8;
  h8 x0 = *(const h8*)p;
  h8 x1 = *(const h8*)(p + 16);
  v16h a;
#pragma unroll
  for (int i = 0; i < 8; ++i) { a[i] = x0[i]; a[i + 8] = x1[i]; }
  return a;
}

// B fragment (32x16 f16, KxN) from B^T stored [N,K]: lane l: n = l&15,
// lanes 0-15 hold K 0-7 & 16-23, lanes 16-31 hold K 8-15 & 24-31.
__device__ inline v16h load_bT_frag(const _Float16* Bt, int ldb, int col0,
                                    int k0, int lane) {
  int n  = lane & 15;
  int kb = (lane >> 4) * 8;
  const _Float16* p = Bt + (size_t)(col0 + n) * ldb + k0 + kb;
  h8 x0 = *(const h8*)p;
  h8 x1 = *(const h8*)(p + 16);
  v16h b;
#pragma unroll
  for (int i = 0; i < 8; ++i) { b[i] = x0[i]; b[i + 8] = x1[i]; }
  return b;
}

// ---------------------------------------------------------------- kernels

// Weight transpose + f32->f16 convert: src [K,N] f32 -> dst [N,K] f16.
__global__ void k_wT(const float* __restrict__ src, _Float16* __restrict__ dst,
                     int K, int N) {
  size_t i = (size_t)blockIdx.x * 256 + threadIdx.x;
  if (i < (size_t)K * N) {
    int k = (int)(i / (unsigned)N);
    int n = (int)(i - (size_t)k * N);
    dst[(size_t)n * K + k] = (_Float16)src[i];
  }
}

// Elementwise f32 -> f16 convert (vectorized: 4 per thread).
__global__ void k_cvt(const float* __restrict__ src, _Float16* __restrict__ dst,
                      size_t n4) {
  size_t i = (size_t)blockIdx.x * 256 + threadIdx.x;
  if (i < n4) {
    float4 v = ((const float4*)src)[i];
    _Float16* d = dst + i * 4;
    d[0] = (_Float16)v.x; d[1] = (_Float16)v.y;
    d[2] = (_Float16)v.z; d[3] = (_Float16)v.w;
  }
}

// LayerNorm over 1024 cols, one block (256 threads) per row, f16 out.
__global__ void k_ln(const float* __restrict__ x, const float* __restrict__ w,
                     const float* __restrict__ b, _Float16* __restrict__ y) {
  int row = blockIdx.x;
  int tid = threadIdx.x;
  const float* xr = x + (size_t)row * 1024;
  float4 v = ((const float4*)xr)[tid];
  float s  = v.x + v.y + v.z + v.w;
  float ss = v.x * v.x + v.y * v.y + v.z * v.z + v.w * v.w;
  __shared__ float rs[256], rss[256];
  rs[tid] = s; rss[tid] = ss;
  __syncthreads();
  for (int st = 128; st > 0; st >>= 1) {
    if (tid < st) { rs[tid] += rs[tid + st]; rss[tid] += rss[tid + st]; }
    __syncthreads();
  }
  float mean = rs[0] * (1.0f / 1024.0f);
  float var  = rss[0] * (1.0f / 1024.0f) - mean * mean;
  float inv  = rsqrtf(var + 1e-5f);
  int c = tid * 4;
  _Float16* yr = y + (size_t)row * 1024;
  yr[c + 0] = (_Float16)((v.x - mean) * inv * w[c + 0] + b[c + 0]);
  yr[c + 1] = (_Float16)((v.y - mean) * inv * w[c + 1] + b[c + 1]);
  yr[c + 2] = (_Float16)((v.z - mean) * inv * w[c + 2] + b[c + 2]);
  yr[c + 3] = (_Float16)((v.w - mean) * inv * w[c + 3] + b[c + 3]);
}

// WMMA GEMM: C = A[M,K] * B (given as B^T [N,K] f16) (+bias).
// One wave per block computes a 64x64 tile (4x4 subtiles of 16x16).
// MODE 0: f16 out scattered to padded [B, HEADS, Spad, DIMH] (Srows rows/batch)
// MODE 1: f32 residual accumulate  out[row,col] += acc + bias
// MODE 2: f16 out = gelu(acc + bias), plain [M,N]
// MODE 3: f16 out scattered TRANSPOSED to padded [B, HEADS, DIMH, Spad]
template <int MODE>
__global__ __launch_bounds__(32) void k_gemm(
    const _Float16* __restrict__ A, const _Float16* __restrict__ Bt,
    const float* __restrict__ bias, void* __restrict__ out, int M, int N, int K,
    int Srows, int Spad) {
  int lane = threadIdx.x;
  int tm = blockIdx.x * 64;
  int tn = blockIdx.y * 64;
  v8f acc[4][4];
#pragma unroll
  for (int i = 0; i < 4; ++i)
#pragma unroll
    for (int j = 0; j < 4; ++j) acc[i][j] = vzero8();

  for (int k0 = 0; k0 < K; k0 += 32) {
    v16h a[4], b[4];
#pragma unroll
    for (int i = 0; i < 4; ++i) a[i] = load_a_frag(A, K, tm + i * 16, k0, lane);
#pragma unroll
    for (int j = 0; j < 4; ++j) b[j] = load_bT_frag(Bt, K, tn + j * 16, k0, lane);
#pragma unroll
    for (int i = 0; i < 4; ++i)
#pragma unroll
      for (int j = 0; j < 4; ++j) acc[i][j] = wmma_f16(a[i], b[j], acc[i][j]);
  }

  int n_ = lane & 15;
  int hi = lane >> 4;
#pragma unroll
  for (int i = 0; i < 4; ++i) {
#pragma unroll
    for (int j = 0; j < 4; ++j) {
      int col = tn + j * 16 + n_;
      float bv = 0.0f;
      if (MODE == 1 || MODE == 2) bv = bias[col];
#pragma unroll
      for (int r = 0; r < 8; ++r) {
        int row = tm + i * 16 + r + hi * 8;
        float val = acc[i][j][r] + bv;
        if (MODE == 1) {
          ((float*)out)[(size_t)row * N + col] += val;
        } else if (MODE == 2) {
          ((_Float16*)out)[(size_t)row * N + col] = (_Float16)gelu_f(val);
        } else {
          unsigned bb = (unsigned)row / (unsigned)Srows;
          unsigned s  = (unsigned)row - bb * (unsigned)Srows;
          int h = col >> 6;
          int d = col & 63;
          size_t idx;
          if (MODE == 0)
            idx = (((size_t)bb * HEADS + h) * Spad + s) * DIMH + d;
          else  // MODE 3: transposed [B,H,DIMH,Spad]
            idx = (((size_t)bb * HEADS + h) * DIMH + d) * Spad + s;
          ((_Float16*)out)[idx] = (_Float16)val;
        }
      }
    }
  }
}

// Attention: one wave per (b, h, 16-row t-tile).
// Q  [B,H,64,64]      f16 (pad rows zero)
// Kp [B,H,Spad,64]    f16 (zero-padded beyond S)   -> B^T layout for Q.K^T
// Vt [B,H,64,Spad]    f16 (V transposed, padded)   -> B^T layout for P.V
// out: [B*Tv, 1024] f16 with column = h*64 + d.
__global__ __launch_bounds__(32) void k_attn(
    const _Float16* __restrict__ Q, const _Float16* __restrict__ Kp,
    const _Float16* __restrict__ Vt, _Float16* __restrict__ out, int S,
    int Spad, int Tv, float scale) {
  __shared__ float sc[16][SPAD_MAX];
  __shared__ __align__(16) _Float16 P[16][SPAD_MAX];
  int lane = threadIdx.x;
  int bid = blockIdx.x;
  int tt = bid & 3;
  int bh = bid >> 2;
  int b  = bh >> 4;
  int h  = bh & 15;
  const _Float16* Qbh = Q  + (size_t)bh * 64 * DIMH;
  const _Float16* Kbh = Kp + (size_t)bh * Spad * DIMH;
  const _Float16* Vbh = Vt + (size_t)bh * DIMH * Spad;

  v16h qa0 = load_a_frag(Qbh, DIMH, tt * 16, 0, lane);
  v16h qa1 = load_a_frag(Qbh, DIMH, tt * 16, 32, lane);
  int n_ = lane & 15;
  int hi = lane >> 4;

  // scores = Q K^T * scale  (K natural [S,64] layout == B^T layout)
  for (int st = 0; st < Spad / 16; ++st) {
    v8f acc = vzero8();
    acc = wmma_f16(qa0, load_bT_frag(Kbh, DIMH, st * 16, 0, lane), acc);
    acc = wmma_f16(qa1, load_bT_frag(Kbh, DIMH, st * 16, 32, lane), acc);
#pragma unroll
    for (int r = 0; r < 8; ++r)
      sc[r + hi * 8][st * 16 + n_] = acc[r] * scale;
  }
  __syncthreads();

  // masked softmax per row (wave cooperates over S)
  for (int r = 0; r < 16; ++r) {
    float mx = -1e30f;
    for (int s = lane; s < S; s += 32) mx = fmaxf(mx, sc[r][s]);
#pragma unroll
    for (int off = 16; off > 0; off >>= 1) mx = fmaxf(mx, __shfl_xor(mx, off, 32));
    float sum = 0.0f;
    for (int s = lane; s < S; s += 32) {
      float e = __expf(sc[r][s] - mx);
      sc[r][s] = e;
      sum += e;
    }
#pragma unroll
    for (int off = 16; off > 0; off >>= 1) sum += __shfl_xor(sum, off, 32);
    float inv = 1.0f / sum;
    for (int s = lane; s < Spad; s += 32)
      P[r][s] = (_Float16)((s < S) ? sc[r][s] * inv : 0.0f);
  }
  __syncthreads();

  // out = P V   (Vt is [DIMH, Spad] per (b,h): exactly the B^T layout)
  for (int nt = 0; nt < 4; ++nt) {
    v8f acc = vzero8();
    for (int k0 = 0; k0 < Spad; k0 += 32) {
      v16h pa = load_a_frag(&P[0][0], SPAD_MAX, 0, k0, lane);
      v16h vb = load_bT_frag(Vbh, Spad, nt * 16, k0, lane);
      acc = wmma_f16(pa, vb, acc);
    }
#pragma unroll
    for (int r = 0; r < 8; ++r) {
      int t = tt * 16 + r + hi * 8;
      if (t < Tv)
        out[((size_t)b * Tv + t) * 1024 + h * DIMH + nt * 16 + n_] =
            (_Float16)acc[r];
    }
  }
}

// ---------------------------------------------------------------- launch

extern "C" void kernel_launch(void* const* d_in, const int* in_sizes, int n_in,
                              void* d_out, int out_size, void* d_ws,
                              size_t ws_size, hipStream_t stream) {
  (void)in_sizes; (void)n_in; (void)out_size; (void)ws_size;

  const float* x    = (const float*)d_in[0];
  const float* img  = (const float*)d_in[1];
  const float* prm  = (const float*)d_in[2];
  const float* ln1w = (const float*)d_in[3];  const float* ln1b = (const float*)d_in[4];
  const float* ln2w = (const float*)d_in[5];  const float* ln2b = (const float*)d_in[6];
  const float* ln3w = (const float*)d_in[7];  const float* ln3b = (const float*)d_in[8];
  const float* ln4w = (const float*)d_in[9];  const float* ln4b = (const float*)d_in[10];
  const float* sawq = (const float*)d_in[11]; const float* sawk = (const float*)d_in[12];
  const float* sawv = (const float*)d_in[13]; const float* sawo = (const float*)d_in[14];
  const float* sabo = (const float*)d_in[15];
  const float* iawq = (const float*)d_in[16]; const float* iawk = (const float*)d_in[17];
  const float* iawv = (const float*)d_in[18]; const float* iawo = (const float*)d_in[19];
  const float* iabo = (const float*)d_in[20];
  const float* pawq = (const float*)d_in[21]; const float* pawk = (const float*)d_in[22];
  const float* pawv = (const float*)d_in[23]; const float* pawo = (const float*)d_in[24];
  const float* pabo = (const float*)d_in[25];
  const float* fcw  = (const float*)d_in[26]; const float* fcb  = (const float*)d_in[27];
  const float* pjw  = (const float*)d_in[28]; const float* pjb  = (const float*)d_in[29];

  float* xo = (float*)d_out;

  const int B = 256, T = 60, D = 1024, BT = 15360;
  const int SIMG = 197, SPIMG = 224, MIMG = 50432;
  const int SPRM = 77,  SPPRM = 96,  MPRM = 19712;
  const int SSELF = 60, SPSELF = 64;

  char* wsp = (char*)d_ws;
  auto alloc = [&](size_t bytes) -> void* {
    void* p = (void*)wsp;
    wsp += (bytes + 255) & ~(size_t)255;
    return p;
  };

  const size_t E1M  = (size_t)1024 * 1024;
  const size_t E768 = (size_t)768 * 1024;
  _Float16* saqT = (_Float16*)alloc(E1M * 2);
  _Float16* sakT = (_Float16*)alloc(E1M * 2);
  _Float16* savT = (_Float16*)alloc(E1M * 2);
  _Float16* saoT = (_Float16*)alloc(E1M * 2);
  _Float16* iaqT = (_Float16*)alloc(E1M * 2);
  _Float16* iakT = (_Float16*)alloc(E768 * 2);
  _Float16* iavT = (_Float16*)alloc(E768 * 2);
  _Float16* iaoT = (_Float16*)alloc(E1M * 2);
  _Float16* paqT = (_Float16*)alloc(E1M * 2);
  _Float16* pakT = (_Float16*)alloc(E768 * 2);
  _Float16* pavT = (_Float16*)alloc(E768 * 2);
  _Float16* paoT = (_Float16*)alloc(E1M * 2);
  _Float16* fcT  = (_Float16*)alloc((size_t)1024 * 4096 * 2);
  _Float16* pjT  = (_Float16*)alloc((size_t)4096 * 1024 * 2);
  _Float16* hln  = (_Float16*)alloc((size_t)BT * D * 2);
  _Float16* imgh = (_Float16*)alloc((size_t)MIMG * 768 * 2);
  _Float16* prmh = (_Float16*)alloc((size_t)MPRM * 768 * 2);
  _Float16* Qb   = (_Float16*)alloc((size_t)B * HEADS * 64 * DIMH * 2);
  _Float16* Kb   = (_Float16*)alloc((size_t)B * HEADS * SPIMG * DIMH * 2);
  _Float16* Vb   = (_Float16*)alloc((size_t)B * HEADS * SPIMG * DIMH * 2);
  _Float16* attb = (_Float16*)alloc((size_t)BT * D * 2);
  _Float16* mid  = Kb;  // MLP hidden reuses Kb+Vb region (contiguous, big enough)

  auto wT = [&](const float* src, _Float16* dst, int K, int N) {
    size_t tot = (size_t)K * N;
    k_wT<<<dim3((unsigned)((tot + 255) / 256)), dim3(256), 0, stream>>>(src, dst, K, N);
  };
  wT(sawq, saqT, 1024, 1024); wT(sawk, sakT, 1024, 1024);
  wT(sawv, savT, 1024, 1024); wT(sawo, saoT, 1024, 1024);
  wT(iawq, iaqT, 1024, 1024); wT(iawk, iakT, 768, 1024);
  wT(iawv, iavT, 768, 1024);  wT(iawo, iaoT, 1024, 1024);
  wT(pawq, paqT, 1024, 1024); wT(pawk, pakT, 768, 1024);
  wT(pawv, pavT, 768, 1024);  wT(pawo, paoT, 1024, 1024);
  wT(fcw,  fcT,  1024, 4096); wT(pjw,  pjT,  4096, 1024);

  // one-shot f32 -> f16 conversion of the attention contexts
  {
    size_t n4 = (size_t)MIMG * 768 / 4;
    k_cvt<<<dim3((unsigned)((n4 + 255) / 256)), dim3(256), 0, stream>>>(img, imgh, n4);
    n4 = (size_t)MPRM * 768 / 4;
    k_cvt<<<dim3((unsigned)((n4 + 255) / 256)), dim3(256), 0, stream>>>(prm, prmh, n4);
  }

  // residual stream lives in d_out (f32)
  hipMemcpyAsync(xo, x, (size_t)BT * D * sizeof(float),
                 hipMemcpyDeviceToDevice, stream);

  auto attn_stage = [&](const float* lw, const float* lb, _Float16* wqT,
                        _Float16* wkT, _Float16* wvT, _Float16* woT,
                        const float* bo, const _Float16* ctx, int Kctx,
                        int Mctx, int S, int Spad) {
    k_ln<<<BT, 256, 0, stream>>>(xo, lw, lb, hln);
    hipMemsetAsync(Qb, 0, (size_t)B * HEADS * 64 * DIMH * 2, stream);
    hipMemsetAsync(Kb, 0, (size_t)B * HEADS * Spad * DIMH * 2, stream);
    hipMemsetAsync(Vb, 0, (size_t)B * HEADS * Spad * DIMH * 2, stream);
    // Q projection (rows per batch = T, pad 64)
    k_gemm<0><<<dim3(BT / 64, D / 64), 32, 0, stream>>>(
        hln, wqT, nullptr, Qb, BT, D, D, T, 64);
    // K projection -> [B,H,Spad,64];  V projection -> transposed [B,H,64,Spad]
    k_gemm<0><<<dim3(Mctx / 64, D / 64), 32, 0, stream>>>(
        ctx, wkT, nullptr, Kb, Mctx, D, Kctx, S, Spad);
    k_gemm<3><<<dim3(Mctx / 64, D / 64), 32, 0, stream>>>(
        ctx, wvT, nullptr, Vb, Mctx, D, Kctx, S, Spad);
    k_attn<<<B * HEADS * 4, 32, 0, stream>>>(Qb, Kb, Vb, attb, S, Spad, T,
                                             0.125f);
    // output projection + residual add into xo
    k_gemm<1><<<dim3(BT / 64, D / 64), 32, 0, stream>>>(
        attb, woT, bo, xo, BT, D, D, 0, 0);
  };

  attn_stage(ln1w, ln1b, saqT, sakT, savT, saoT, sabo, hln, 1024, BT,
             SSELF, SPSELF);
  attn_stage(ln2w, ln2b, iaqT, iakT, iavT, iaoT, iabo, imgh, 768, MIMG, SIMG,
             SPIMG);
  attn_stage(ln3w, ln3b, paqT, pakT, pavT, paoT, pabo, prmh, 768, MPRM, SPRM,
             SPPRM);

  // MLP
  k_ln<<<BT, 256, 0, stream>>>(xo, ln4w, ln4b, hln);
  k_gemm<2><<<dim3(BT / 64, 4096 / 64), 32, 0, stream>>>(
      hln, fcT, fcb, mid, BT, 4096, 1024, 0, 0);
  k_gemm<1><<<dim3(BT / 64, D / 64), 32, 0, stream>>>(
      mid, pjT, pjb, xo, BT, D, 4096, 0, 0);
}
